// PairwiseCosineSimilarity_72284299592196
// MI455X (gfx1250) — matrix operand order repaired and checked by reading, not compiled, not run on hardware
//
#include <hip/hip_runtime.h>

typedef __attribute__((ext_vector_type(16))) __bf16 v16bf;
typedef __attribute__((ext_vector_type(8)))  float  v8f;

#define BATCH 4
#define MDIM  4096
#define NDIM  4096
#define DDIM  256
#define KCHUNKS 8          // 256 / 32
#define TILES_PER_DIM 256  // 4096 / 16
#define EPS 1e-8f

// ---------------------------------------------------------------------------
// Row L2 norms: one wave (32 lanes) per row of 256 floats.
// ---------------------------------------------------------------------------
__global__ void norm_kernel(const float* __restrict__ src,
                            float* __restrict__ norms, int nrows) {
  const int lane = threadIdx.x & 31;
  const int wave = threadIdx.x >> 5;
  const int row  = blockIdx.x * 8 + wave;
  if (row >= nrows) return;
  const float* p = src + (size_t)row * DDIM;
  float s = 0.f;
  #pragma unroll
  for (int d = lane; d < DDIM; d += 32) { float v = p[d]; s += v * v; }
  #pragma unroll
  for (int off = 16; off > 0; off >>= 1) s += __shfl_xor(s, off, 32);
  if (lane == 0) norms[row] = sqrtf(s);
}

// ---------------------------------------------------------------------------
// f32 -> (bf16_hi, bf16_lo) split, written directly in WMMA fragment layout.
//   A layout (16x32, ISA 7.12.2): lane 0-15 -> M=lane, K = j + (j<8?0:8);
//                                 lane 16-31 -> +8 on K.
//   B layout (32x16): lane&15 -> N, K = j + (j<8?0:8) + (lane<16?0:8).
// One thread per fragment-lane: converts 16 elements, stores 16 hi + 16 lo.
// Fragment array index == flat thread id: ((b*256+tile)*8+kc)*32+lane.
// ---------------------------------------------------------------------------
__device__ __forceinline__ unsigned short bf16_rne(float f) {
  unsigned u = __float_as_uint(f);
  unsigned r = u + 0x7FFFu + ((u >> 16) & 1u);
  return (unsigned short)(r >> 16);
}

__global__ void convert_kernel(const float* __restrict__ src,
                               unsigned short* __restrict__ hi,
                               unsigned short* __restrict__ lo,
                               int b_layout) {
  const int t    = blockIdx.x * blockDim.x + threadIdx.x; // < 4*256*8*32
  const int lane = t & 31;
  const int kc   = (t >> 5) & 7;
  const int tile = (t >> 8) & 255;
  const int b    = t >> 16;
  const int row  = lane & 15;

  const float* s = src + (((size_t)b * MDIM + tile * 16 + row) * DDIM) + kc * 32;

  unsigned short hv[16], lv[16];
  #pragma unroll
  for (int j = 0; j < 16; ++j) {
    const int K = b_layout ? (j + ((j < 8) ? 0 : 8) + ((lane < 16) ? 0 : 8))
                           : (j + ((j < 8) ? 0 : 8) + ((lane < 16) ? 0 : 8));
    const float f = s[K];
    const unsigned short h = bf16_rne(f);
    const float hf = __uint_as_float(((unsigned)h) << 16);
    hv[j] = h;
    lv[j] = bf16_rne(f - hf);
  }
  const size_t base = (size_t)t * 16;
  #pragma unroll
  for (int j = 0; j < 16; ++j) { hi[base + j] = hv[j]; lo[base + j] = lv[j]; }
}

// ---------------------------------------------------------------------------
// Batched GEMM + cosine normalization.
// 256 threads = 8 waves; waves in 4(m) x 2(n) grid; wave tile = 32x64
// (2 m-subtiles x 4 n-subtiles of 16x16). Workgroup tile = 128x128.
// 3 bf16 WMMAs per (subtile, K-chunk): hi*hi + hi*lo + lo*hi  (~f32 accuracy).
// ---------------------------------------------------------------------------
__global__ __launch_bounds__(256, 1)
void gemm_kernel(const unsigned short* __restrict__ Ahi,
                 const unsigned short* __restrict__ Alo,
                 const unsigned short* __restrict__ Bhi,
                 const unsigned short* __restrict__ Blo,
                 const float* __restrict__ xn,
                 const float* __restrict__ yn,
                 float* __restrict__ out) {
  const int lane = threadIdx.x & 31;
  const int wave = threadIdx.x >> 5;
  const int wm   = wave & 3;   // 4 waves along M
  const int wn   = wave >> 2;  // 2 waves along N
  const int b    = blockIdx.z;
  const int mt0  = blockIdx.y * 8 + wm * 2;  // first of 2 m-subtiles (16-row tiles)
  const int nt0  = blockIdx.x * 8 + wn * 4;  // first of 4 n-subtiles (16-col tiles)

  v8f acc[2][4];
  #pragma unroll
  for (int mi = 0; mi < 2; ++mi)
    #pragma unroll
    for (int ni = 0; ni < 4; ++ni) acc[mi][ni] = {};

  for (int kc = 0; kc < KCHUNKS; ++kc) {
    v16bf ah[2], al[2], bh[4], bl[4];
    #pragma unroll
    for (int mi = 0; mi < 2; ++mi) {
      const size_t idx =
          ((((size_t)b * TILES_PER_DIM + (mt0 + mi)) * KCHUNKS + kc) * 32 + lane) * 16;
      ah[mi] = *(const v16bf*)(Ahi + idx);
      al[mi] = *(const v16bf*)(Alo + idx);
    }
    #pragma unroll
    for (int ni = 0; ni < 4; ++ni) {
      const size_t idx =
          ((((size_t)b * TILES_PER_DIM + (nt0 + ni)) * KCHUNKS + kc) * 32 + lane) * 16;
      bh[ni] = *(const v16bf*)(Bhi + idx);
      bl[ni] = *(const v16bf*)(Blo + idx);
    }
    #pragma unroll
    for (int mi = 0; mi < 2; ++mi) {
      #pragma unroll
      for (int ni = 0; ni < 4; ++ni) {
        acc[mi][ni] = __builtin_amdgcn_wmma_f32_16x16x32_bf16(
            false, ah[mi], false, bh[ni], (short)0, acc[mi][ni], false, false);
        acc[mi][ni] = __builtin_amdgcn_wmma_f32_16x16x32_bf16(
            false, ah[mi], false, bl[ni], (short)0, acc[mi][ni], false, false);
        acc[mi][ni] = __builtin_amdgcn_wmma_f32_16x16x32_bf16(
            false, al[mi], false, bh[ni], (short)0, acc[mi][ni], false, false);
      }
    }
  }

  // Epilogue: C layout -> lane 0-15: N=lane, M=e; lane 16-31: N=lane-16, M=e+8.
  const int ncol   = lane & 15;
  const int rowoff = (lane >> 4) << 3;
  #pragma unroll
  for (int mi = 0; mi < 2; ++mi) {
    #pragma unroll
    for (int e = 0; e < 8; ++e) {
      const int m = (mt0 + mi) * 16 + e + rowoff;
      const float ynv = yn[b * MDIM + m];
      #pragma unroll
      for (int ni = 0; ni < 4; ++ni) {
        const int n = (nt0 + ni) * 16 + ncol;
        float denom = ynv * xn[b * NDIM + n];
        denom = fmaxf(denom, EPS);
        out[((size_t)b * MDIM + m) * NDIM + n] = acc[mi][ni][e] / denom;
      }
    }
  }
}

// ---------------------------------------------------------------------------
extern "C" void kernel_launch(void* const* d_in, const int* in_sizes, int n_in,
                              void* d_out, int out_size, void* d_ws, size_t ws_size,
                              hipStream_t stream) {
  const float* x = (const float*)d_in[0];  // [4,4096,256] -> B operand (N rows)
  const float* y = (const float*)d_in[1];  // [4,4096,256] -> A operand (M rows)
  float* out = (float*)d_out;

  char* ws = (char*)d_ws;
  const size_t FRAG_BYTES = (size_t)BATCH * MDIM * DDIM * sizeof(unsigned short); // 8 MB
  unsigned short* Ahi = (unsigned short*)(ws);
  unsigned short* Alo = (unsigned short*)(ws + 1 * FRAG_BYTES);
  unsigned short* Bhi = (unsigned short*)(ws + 2 * FRAG_BYTES);
  unsigned short* Blo = (unsigned short*)(ws + 3 * FRAG_BYTES);
  float* ynorm = (float*)(ws + 4 * FRAG_BYTES);
  float* xnorm = (float*)(ws + 4 * FRAG_BYTES + (size_t)BATCH * MDIM * sizeof(float));

  const int nrows = BATCH * MDIM;                    // 16384
  norm_kernel<<<nrows / 8, 256, 0, stream>>>(x, xnorm, nrows);
  norm_kernel<<<nrows / 8, 256, 0, stream>>>(y, ynorm, nrows);

  const int frag_threads = BATCH * TILES_PER_DIM * KCHUNKS * 32; // 262144
  convert_kernel<<<frag_threads / 256, 256, 0, stream>>>(y, Ahi, Alo, 0); // A layout
  convert_kernel<<<frag_threads / 256, 256, 0, stream>>>(x, Bhi, Blo, 1); // B layout

  dim3 grid(NDIM / 128, MDIM / 128, BATCH);          // (32, 32, 4)
  gemm_kernel<<<grid, 256, 0, stream>>>(Ahi, Alo, Bhi, Blo, xnorm, ynorm, out);
}
